// ExactCascade2D_10625749090839
// MI455X (gfx1250) — matrix operand order repaired and checked by reading, not compiled, not run on hardware
//
#include <hip/hip_runtime.h>
#include <hip/hip_bf16.h>
#include <math.h>

#define B_BATCH 512
#define L1C 48
#define L2C 48
#define DIM 2304            // 48*48
#define NPAIR 16            // 4x4 transition matrices
#define KBIG (NPAIR * DIM)  // 36864
#define NCHUNK (DIM / 32)   // 72 K-chunks per d-pass
#define NDEPTH 4
#define TWO_PI_F 6.2831853071795864f

// Byte strides of the swizzled bf16 T layout: [dchunk][nb][pair][half][256]
#define FRAG_BYTES   1024                      // one (nb,pair) fragment
#define NB_BYTES     (16 * FRAG_BYTES)         // 16 pairs          = 16 KB
#define CHUNK_BYTES  (144 * NB_BYTES)          // 144 nb blocks     = 2.25 MB

typedef __attribute__((ext_vector_type(16))) __bf16 v16bf;
typedef __attribute__((ext_vector_type(8)))  float  v8f;

__device__ __forceinline__ unsigned short f2bf_rne(float f) {
    unsigned int u = __float_as_uint(f);
    unsigned int r = u + 0x7fffu + ((u >> 16) & 1u);
    return (unsigned short)(r >> 16);
}

// ---------------------------------------------------------------------------
// Convert T (fp32, [16 matrices each DIM x DIM] = [KBIG x DIM]) into bf16,
// swizzled so one K-chunk's fragments are contiguous:
//   k -> kc=k/32 -> pair=kc/72, dchunk=kc%72 ; half=(k%32)/16, kin=k%16
//   e -> nb=e/16, nc=e%16
//   off = ((((dchunk*144 + nb)*16 + pair)*2 + half)*256) + nc*16 + kin
// Fragment (dchunk, nb, pair) = 1024 contiguous bytes, in this kernel's
// 16-bit B lane layout (lane = column, two K-halves split at lane 16).
// ---------------------------------------------------------------------------
__global__ void convert_T_kernel(const float* __restrict__ T,
                                 unsigned short* __restrict__ Tb) {
    const long long total  = (long long)KBIG * DIM;
    const long long stride = (long long)gridDim.x * blockDim.x;
    for (long long i = (long long)blockIdx.x * blockDim.x + threadIdx.x;
         i < total; i += stride) {
        int       e  = (int)(i % DIM);
        long long k  = i / DIM;
        int kc     = (int)(k >> 5);
        int kr     = (int)(k & 31);
        int pair   = kc / NCHUNK;
        int dchunk = kc % NCHUNK;
        int half   = kr >> 4;
        int kin    = kr & 15;
        int nb     = e >> 4;
        int nc     = e & 15;
        long long off =
            ((((long long)(dchunk * 144 + nb) * 16 + pair) * 2 + half) * 256)
            + nc * 16 + kin;
        Tb[off] = f2bf_rne(T[i]);
    }
}

// Phi0 = one_hot(clip(floor(x),0,47)*48 + clip(floor(y),0,47))
__global__ void init_phi_kernel(const float* __restrict__ x,
                                const float* __restrict__ y,
                                float* __restrict__ Phi) {
    int i = blockIdx.x * blockDim.x + threadIdx.x;
    if (i >= B_BATCH * DIM) return;
    int b = i / DIM, d = i % DIM;
    int a = (int)floorf(x[b]); a = a < 0 ? 0 : (a > L1C - 1 ? L1C - 1 : a);
    int c = (int)floorf(y[b]); c = c < 0 ? 0 : (c > L2C - 1 ? L2C - 1 : c);
    Phi[i] = (d == a * L2C + c) ? 1.0f : 0.0f;
}

__global__ void init_z_kernel(const float* __restrict__ x,
                              const float* __restrict__ y,
                              float* __restrict__ z) {
    int b = blockIdx.x * blockDim.x + threadIdx.x;
    if (b >= B_BATCH) return;
    float u = x[b] - floorf(x[b]);
    float v = y[b] - floorf(y[b]);
    z[b * 4 + 0] = cosf(TWO_PI_F * u);
    z[b * 4 + 1] = sinf(TWO_PI_F * u);
    z[b * 4 + 2] = cosf(TWO_PI_F * v);
    z[b * 4 + 3] = sinf(TWO_PI_F * v);
}

// Per-step controller: chix = softmax(z@Wsel[:,:4]); chiy = softmax(z@Wsel[:,4:]);
// then z <- tanh(z @ W_ctrl + b_ctrl).
__global__ void chi_kernel(float* __restrict__ z,
                           const float* __restrict__ W_sel,
                           const float* __restrict__ W_ctrl,
                           const float* __restrict__ b_ctrl,
                           float* __restrict__ chix,
                           float* __restrict__ chiy) {
    int b = blockIdx.x * blockDim.x + threadIdx.x;
    if (b >= B_BATCH) return;
    float zb[4];
    for (int j = 0; j < 4; j++) zb[j] = z[b * 4 + j];
    float lg[8];
    for (int c = 0; c < 8; c++) {
        float s = 0.f;
        for (int j = 0; j < 4; j++) s += zb[j] * W_sel[j * 8 + c];
        lg[c] = s;
    }
    float cx[4], cy[4];
    float mx = fmaxf(fmaxf(lg[0], lg[1]), fmaxf(lg[2], lg[3]));
    float sx = 0.f;
    for (int q = 0; q < 4; q++) { cx[q] = expf(lg[q] - mx); sx += cx[q]; }
    for (int q = 0; q < 4; q++) chix[b * 4 + q] = cx[q] / sx;
    float my = fmaxf(fmaxf(lg[4], lg[5]), fmaxf(lg[6], lg[7]));
    float sy = 0.f;
    for (int p = 0; p < 4; p++) { cy[p] = expf(lg[4 + p] - my); sy += cy[p]; }
    for (int p = 0; p < 4; p++) chiy[b * 4 + p] = cy[p] / sy;
    float zn[4];
    for (int c = 0; c < 4; c++) {
        float s = b_ctrl[c];
        for (int j = 0; j < 4; j++) s += zb[j] * W_ctrl[j * 4 + c];
        zn[c] = tanhf(s);
    }
    for (int j = 0; j < 4; j++) z[b * 4 + j] = zn[j];
}

// ---------------------------------------------------------------------------
// Build the chi_x-scaled bf16 A operand in fragment-ready layout:
//   Ab byte offset ((q*512 + b)*72 + d)*64 + half*32, 16 bf16 per half,
//   element j <- chix[b][q] * Phi[b][d*32 + half*16 + j]
// Contiguous k order (k = 16*half + j), the same order the T swizzle gives B.
// ---------------------------------------------------------------------------
__global__ void prep_A_kernel(const float* __restrict__ Phi,
                              const float* __restrict__ chix,
                              unsigned short* __restrict__ Ab) {
    int i = blockIdx.x * blockDim.x + threadIdx.x;  // over 4*512*2304
    if (i >= 4 * B_BATCH * DIM) return;
    int j    = i & 15;
    int half = (i >> 4) & 1;
    int rest = i >> 5;               // (q*512 + b)*72 + d
    int d    = rest % NCHUNK;
    int qb   = rest / NCHUNK;
    int b    = qb & (B_BATCH - 1);
    int q    = qb >> 9;
    int K    = d * 32 + half * 16 + j;
    float v  = chix[b * 4 + q] * Phi[(long long)b * DIM + K];
    Ab[i] = f2bf_rne(v);
}

// ---------------------------------------------------------------------------
// Cascade GEMM step, pure WMMA dataflow (A and B both pre-swizzled bf16):
//   PhiOut[b,e] = sum_p chiy[b][p] * sum_q (A_q[b,:] @ T[q,p])[e]
// Wave tile 16(M) x 32(N): acc[p*2+t] (8 x v8f). Block: 8 waves 4(M) x 2(N)
// -> 64 x 64 tile, grid (36, 8). Per K-chunk: 32 WMMAs, B through an 8-deep
// register pipeline, A double-buffered one chunk ahead (static ping-pong via
// 2x-unrolled chunk loop). All loads use immediate offsets off bumped bases.
// ---------------------------------------------------------------------------
__global__ __launch_bounds__(256) void cascade_gemm_kernel(
        const unsigned short* __restrict__ Ab,
        const unsigned short* __restrict__ Tb,
        const float* __restrict__ chiy,
        float* __restrict__ PhiOut) {
    const int lane    = threadIdx.x & 31;
    const int wave    = threadIdx.x >> 5;
    const int wm      = wave >> 1;   // 0..3
    const int wn      = wave & 1;    // 0..1
    const int m0      = blockIdx.y * 64 + wm * 16;
    const int n0      = blockIdx.x * 64 + wn * 32;
    const int row     = m0 + (lane & 15);
    const int halfsel = (lane < 16) ? 0 : 1;
    const int nb0     = n0 >> 4;

    v8f acc[8] = {};   // acc[p*2+t]

    const char* TbB = (const char*)Tb;
    const char* AbB = (const char*)Ab;

    unsigned vbase = (unsigned)nb0 * NB_BYTES
                   + (unsigned)(halfsel * 512 + (lane & 15) * 32);
    unsigned abase = (unsigned)row * (NCHUNK * 64) + (unsigned)(halfsel * 32);

#define AQ_IMM(q)   ((unsigned)(q) * (unsigned)(B_BATCH * NCHUNK * 64))
#define FRAG_IMM(i) ((unsigned)(((i) & 1) * NB_BYTES + \
                     ((((i) >> 3) * 4 + (((i) >> 1) & 3)) * FRAG_BYTES)))

    // ---- prologue: A fragments for chunk 0, first 8 B fragments ----
    v16bf afA[4], afB[4];
#pragma unroll
    for (int q = 0; q < 4; q++)
        afA[q] = *reinterpret_cast<const v16bf*>(AbB + abase + AQ_IMM(q));

    v16bf buf[8];
#pragma unroll
    for (int j = 0; j < 8; j++)
        buf[j] = *reinterpret_cast<const v16bf*>(TbB + vbase + FRAG_IMM(j));

    // One K-chunk: issue A for the next chunk, stream B 8 ahead, 32 WMMAs.
#define CHUNK_BODY(AFU, AFL, LOAD_A, PREFETCH_B)                            \
    {                                                                       \
        if (LOAD_A) {                                                       \
            _Pragma("unroll")                                               \
            for (int q = 0; q < 4; q++)                                     \
                AFL[q] = *reinterpret_cast<const v16bf*>(                   \
                    AbB + (abase + 64u) + AQ_IMM(q));                       \
        }                                                                   \
        const unsigned vnext = vbase + CHUNK_BYTES;                         \
        _Pragma("unroll")                                                   \
        for (int i = 0; i < 32; ++i) {                                      \
            v16bf bcur = buf[i & 7];                                        \
            if (i < 24)                                                     \
                buf[i & 7] = *reinterpret_cast<const v16bf*>(               \
                    TbB + vbase + FRAG_IMM(i + 8));                         \
            else if (PREFETCH_B)                                            \
                buf[i & 7] = *reinterpret_cast<const v16bf*>(               \
                    TbB + vnext + FRAG_IMM(i - 24));                        \
            acc[i & 7] = __builtin_amdgcn_wmma_f32_16x16x32_bf16(           \
                false, AFU[i >> 3], false, bcur, (short)0, acc[i & 7],      \
                false, false);                                              \
        }                                                                   \
        vbase = vnext;                                                      \
        abase += 64u;                                                       \
    }

    // chunks 0..69 in static ping-pong pairs (branch-free bodies)
    for (int dd = 0; dd < (NCHUNK - 2) / 2; ++dd) {
        CHUNK_BODY(afA, afB, true, true);
        CHUNK_BODY(afB, afA, true, true);
    }
    // chunk 70: uses afA, loads afB (chunk 71), B rolls into chunk 71
    CHUNK_BODY(afA, afB, true, true);
    // chunk 71: uses afB, nothing further to fetch
    CHUNK_BODY(afB, afA, false, false);

#undef CHUNK_BODY
#undef FRAG_IMM
#undef AQ_IMM

    // C/D layout: N = lane%16 (+t*16), M = vgpr index i (+8 for lanes 16..31).
    const int colbase = n0 + (lane & 15);
    const int rbase   = m0 + halfsel * 8;
#pragma unroll
    for (int i = 0; i < 8; i++) {
        const int r = rbase + i;
        float cy[4];
#pragma unroll
        for (int p = 0; p < 4; p++) cy[p] = chiy[r * 4 + p];
#pragma unroll
        for (int t = 0; t < 2; t++) {
            float v = cy[0] * acc[0 * 2 + t][i];
            v = fmaf(cy[1], acc[1 * 2 + t][i], v);
            v = fmaf(cy[2], acc[2 * 2 + t][i], v);
            v = fmaf(cy[3], acc[3 * 2 + t][i], v);
            PhiOut[(long long)r * DIM + colbase + t * 16] = v;
        }
    }
}

// Final readout: reads = z@W_read + b_read; 4 sigmoid atoms; exact-min gadget.
__global__ void final_kernel(const float* __restrict__ z,
                             const float* __restrict__ W_read,
                             const float* __restrict__ b_read,
                             const float* __restrict__ w_atom,
                             const float* __restrict__ b_atom,
                             const float* __restrict__ Phi,
                             float* __restrict__ out) {
    int b = blockIdx.x * blockDim.x + threadIdx.x;
    if (b >= B_BATCH) return;
    float zb[4];
    for (int j = 0; j < 4; j++) zb[j] = z[b * 4 + j];
    float r[8];
    for (int c = 0; c < 8; c++) {
        float s = b_read[c];
        for (int j = 0; j < 4; j++) s += zb[j] * W_read[j * 8 + c];
        r[c] = s;
    }
    float w0 = w_atom[0], w1 = w_atom[1], ba = b_atom[0];
    float bv[4];
    for (int i = 0; i < 4; i++) {
        float t = r[2 * i] * w0 + r[2 * i + 1] * w1 + ba;
        bv[i] = 1.f / (1.f + expf(-t));
    }
    float m01 = 0.5f * (bv[0] + bv[1] - fabsf(bv[0] - bv[1]));
    float m23 = 0.5f * (bv[2] + bv[3] - fabsf(bv[2] - bv[3]));
    float sf  = 0.5f * (m01 + m23 - fabsf(m01 - m23));
    out[b] = Phi[(long long)b * DIM] * sf;
}

extern "C" void kernel_launch(void* const* d_in, const int* in_sizes, int n_in,
                              void* d_out, int out_size, void* d_ws, size_t ws_size,
                              hipStream_t stream) {
    const float* x      = (const float*)d_in[0];
    const float* y      = (const float*)d_in[1];
    const float* T      = (const float*)d_in[2];
    const float* W_sel  = (const float*)d_in[3];
    const float* W_ctrl = (const float*)d_in[4];
    const float* b_ctrl = (const float*)d_in[5];
    const float* W_read = (const float*)d_in[6];
    const float* b_read = (const float*)d_in[7];
    const float* w_atom = (const float*)d_in[8];
    const float* b_atom = (const float*)d_in[9];
    float* out = (float*)d_out;

    char* ws = (char*)d_ws;
    size_t szT      = (size_t)KBIG * DIM * sizeof(unsigned short);        // ~170 MB
    size_t szPhi    = (size_t)B_BATCH * DIM * sizeof(float);              // ~4.7 MB
    size_t szAb     = (size_t)4 * B_BATCH * DIM * sizeof(unsigned short); // ~9.4 MB
    size_t offT     = 0;
    size_t offPhiA  = (offT + szT + 255) & ~(size_t)255;
    size_t offPhiB  = (offPhiA + szPhi + 255) & ~(size_t)255;
    size_t offAb    = (offPhiB + szPhi + 255) & ~(size_t)255;
    size_t offZ     = (offAb + szAb + 255) & ~(size_t)255;
    size_t offChiX  = (offZ + (size_t)B_BATCH * 4 * sizeof(float) + 255) & ~(size_t)255;
    size_t offChiY  = (offChiX + (size_t)B_BATCH * 4 * sizeof(float) + 255) & ~(size_t)255;

    unsigned short* Tb  = (unsigned short*)(ws + offT);
    float* PhiA  = (float*)(ws + offPhiA);
    float* PhiB  = (float*)(ws + offPhiB);
    unsigned short* Abuf = (unsigned short*)(ws + offAb);
    float* zbuf  = (float*)(ws + offZ);
    float* chixb = (float*)(ws + offChiX);
    float* chiyb = (float*)(ws + offChiY);

    convert_T_kernel<<<8192, 256, 0, stream>>>(T, Tb);
    init_phi_kernel<<<(B_BATCH * DIM + 255) / 256, 256, 0, stream>>>(x, y, PhiA);
    init_z_kernel<<<2, 256, 0, stream>>>(x, y, zbuf);

    float* pin  = PhiA;
    float* pout = PhiB;
    for (int step = 0; step < NDEPTH; ++step) {
        chi_kernel<<<2, 256, 0, stream>>>(zbuf, W_sel, W_ctrl, b_ctrl,
                                          chixb, chiyb);
        prep_A_kernel<<<(4 * B_BATCH * DIM + 255) / 256, 256, 0, stream>>>(
            pin, chixb, Abuf);
        dim3 grid(DIM / 64, B_BATCH / 64);
        cascade_gemm_kernel<<<grid, 256, 0, stream>>>(Abuf, Tb, chiyb, pout);
        float* t = pin; pin = pout; pout = t;
    }
    final_kernel<<<2, 256, 0, stream>>>(zbuf, W_read, b_read, w_atom, b_atom,
                                        pin, out);
}
